// FlashMultiHeadAttention_6073083756832
// MI455X (gfx1250) — matrix operand order repaired and checked by reading, not compiled, not run on hardware
//
#include <hip/hip_runtime.h>
#include <hip/hip_bf16.h>

// ---------------------------------------------------------------------------
// Fused QKV+RoPE projection (f16 WMMA GEMM) + causal flash attention (f16 WMMA)
// for MI455X / gfx1250 (wave32, 16x16x32 f16 WMMA, f32 accumulate).
// ---------------------------------------------------------------------------

typedef __attribute__((ext_vector_type(16))) _Float16 f16x16;
typedef __attribute__((ext_vector_type(8)))  _Float16 f16x8;
typedef __attribute__((ext_vector_type(4)))  _Float16 f16x4;
typedef __attribute__((ext_vector_type(8)))  float    f32x8;
typedef __attribute__((ext_vector_type(4)))  float    f32x4;

#define NTOK 4096      // B*S
#define DIM  1024
#define SEQ  1024
#define NH   16
#define HD   64
#define LSTR 40        // halves per padded LDS row (80 bytes -> conflict-free, 16B aligned)

__device__ __forceinline__ f16x16 cat8(f16x8 lo, f16x8 hi) {
  return __builtin_shufflevector(lo, hi, 0,1,2,3,4,5,6,7,8,9,10,11,12,13,14,15);
}

// ---------------------------------------------------------------------------
// Kernel 1: qkv = hidden @ Wqkv^T + b ; RoPE(q,k) ; store q,k as [b][h][s][d]
// f16 and v transposed as [b][h][d][s] f16.
// Block: 256 threads (8 waves). Tile: 128 tokens x 128 out-cols, K-step 32.
// Wave w: rows 32*(w&3).., cols 64*(w>>2).. -> 2x4 grid of 16x16 WMMA tiles.
// ---------------------------------------------------------------------------
__global__ __launch_bounds__(256) void qkv_rope_kernel(
    const float* __restrict__ hidden, const float* __restrict__ Wqkv,
    const float* __restrict__ bias,
    _Float16* __restrict__ qd, _Float16* __restrict__ kd, _Float16* __restrict__ vTd)
{
  __shared__ __align__(16) _Float16 ldsA[128 * LSTR];
  __shared__ __align__(16) _Float16 ldsB[128 * LSTR];

  const int tid   = threadIdx.x;
  const int lane  = tid & 31;
  const int wave  = tid >> 5;
  const int lhalf = lane >> 4;   // 0 | 1
  const int lmod  = lane & 15;
  const int m0 = blockIdx.y * 128;          // token tile base
  const int n0 = blockIdx.x * 128;          // out-col tile base
  const int wm = 32 * (wave & 3);           // wave row offset within tile
  const int wn = 64 * (wave >> 2);          // wave col offset within tile

  f32x8 acc[2][4];
#pragma unroll
  for (int i = 0; i < 2; ++i)
#pragma unroll
    for (int j = 0; j < 4; ++j) acc[i][j] = f32x8{0,0,0,0,0,0,0,0};

  for (int k0 = 0; k0 < DIM; k0 += 32) {
    // cooperative load + f32->f16 convert into LDS (128x32 each)
#pragma unroll
    for (int i = 0; i < 4; ++i) {
      const int linear = tid + 256 * i;
      const int row = linear >> 3;
      const int col = (linear & 7) << 2;
      f32x4 a = *(const f32x4*)(hidden + (size_t)(m0 + row) * DIM + k0 + col);
      f16x4 ha = { (_Float16)a[0], (_Float16)a[1], (_Float16)a[2], (_Float16)a[3] };
      *(f16x4*)(&ldsA[row * LSTR + col]) = ha;
      f32x4 b = *(const f32x4*)(Wqkv + (size_t)(n0 + row) * DIM + k0 + col);
      f16x4 hb = { (_Float16)b[0], (_Float16)b[1], (_Float16)b[2], (_Float16)b[3] };
      *(f16x4*)(&ldsB[row * LSTR + col]) = hb;
    }
    if (k0 + 32 < DIM) {   // prefetch next K slab (global_prefetch_b8)
      __builtin_prefetch(hidden + (size_t)(m0 + (tid >> 3)) * DIM + k0 + 32, 0, 1);
      __builtin_prefetch(Wqkv   + (size_t)(n0 + (tid >> 3)) * DIM + k0 + 32, 0, 1);
    }
    __syncthreads();

    // A fragment (16x32 f16): lane holds row M=lmod; halves[0:8]=K(8*lhalf..),
    // halves[8:16]=K(16+8*lhalf..)
    f16x16 afrag[2];
#pragma unroll
    for (int mt = 0; mt < 2; ++mt) {
      const _Float16* p = &ldsA[(wm + 16 * mt + lmod) * LSTR + 8 * lhalf];
      afrag[mt] = cat8(*(const f16x8*)p, *(const f16x8*)(p + 16));
    }
    // B fragment (32x16 f16): lane holds col N=lmod; halves[0:8]=K(8*lhalf..), halves[8:16]=K(16+8*lhalf..)
    f16x16 bfrag[4];
#pragma unroll
    for (int nt = 0; nt < 4; ++nt) {
      const _Float16* p = &ldsB[(wn + 16 * nt + lmod) * LSTR + 8 * lhalf];
      bfrag[nt] = cat8(*(const f16x8*)p, *(const f16x8*)(p + 16));
    }
#pragma unroll
    for (int mt = 0; mt < 2; ++mt)
#pragma unroll
      for (int nt = 0; nt < 4; ++nt)
        acc[mt][nt] = __builtin_amdgcn_wmma_f32_16x16x32_f16(
            false, afrag[mt], false, bfrag[nt], (short)0, acc[mt][nt], false, false);
    __syncthreads();
  }

  // ------------------ epilogue: bias + RoPE + scatter stores ------------------
  const int cb = n0 + wn;                 // wave col base, multiple of 64 (head-aligned)
  const int kind = (cb < 1024) ? 0 : ((cb < 2048) ? 1 : 2);   // q | k | v

#pragma unroll
  for (int mt = 0; mt < 2; ++mt) {
#pragma unroll
    for (int r = 0; r < 8; ++r) {
      const int m = m0 + wm + 16 * mt + 8 * lhalf + r;  // global token row
      const int b = m >> 10;
      const int s = m & 1023;
      if (kind == 2) {                    // V: bias only, store transposed [b][h][d][s]
#pragma unroll
        for (int t = 0; t < 4; ++t) {
          const int c = cb + 16 * t + lmod;
          const float v = acc[mt][t][r] + bias[c];
          const int cc = c - 2048;
          const int h = cc >> 6, d = cc & 63;
          vTd[(((size_t)b * NH + h) * HD + d) * SEQ + s] = (_Float16)v;
        }
      } else {                            // Q/K: bias + RoPE (pair cols j, j+32)
        _Float16* dst = (kind == 0) ? qd : kd;
        const int coff = (kind == 0) ? 0 : 1024;
#pragma unroll
        for (int t = 0; t < 2; ++t) {
          const int c  = cb + 16 * t + lmod;          // first-half column
          const float x0 = acc[mt][t][r]     + bias[c];
          const float x1 = acc[mt][t + 2][r] + bias[c + 32];
          const int cc = c - coff;
          const int h = cc >> 6;
          const int j = cc & 63;                      // 0..31 (first half of head)
          // inv_freq = 10000^(-2j/64); same angle for j and j+32
          const float invf = __expf(-(float)(2 * j) * (9.210340371976184f / 64.0f));
          const float th = (float)s * invf;
          float sth, cth;
          __sincosf(th, &sth, &cth);
          const float o0 = x0 * cth - x1 * sth;       // out[j]    = x0*cos - x1*sin
          const float o1 = x1 * cth + x0 * sth;       // out[j+32] = x1*cos + x0*sin
          const size_t base = (((size_t)b * NH + h) * SEQ + s) * HD;
          dst[base + j]      = (_Float16)o0;
          dst[base + j + 32] = (_Float16)o1;
        }
      }
    }
  }
}

// ---------------------------------------------------------------------------
// Kernel 2: causal flash attention. Block: 128 threads (4 waves), one (b,h) and
// 64 q-rows per block; wave w owns 16 q-rows. kv tiles of 32.
// All q/k/vT reads are L2-resident (24MB << 192MB L2).
// ---------------------------------------------------------------------------
__global__ __launch_bounds__(128) void flash_attn_kernel(
    const _Float16* __restrict__ qd, const _Float16* __restrict__ kd,
    const _Float16* __restrict__ vTd, float* __restrict__ out)
{
  __shared__ __align__(16) _Float16 plds[4 * 16 * LSTR];  // per-wave 16x32 P scratch

  const int tid   = threadIdx.x;
  const int lane  = tid & 31;
  const int wave  = tid >> 5;
  const int lhalf = lane >> 4;
  const int lmod  = lane & 15;
  const int bh = blockIdx.x;                  // b*NH + h
  const int b  = bh >> 4, h = bh & 15;
  const int q0 = blockIdx.y * 64 + wave * 16; // wave's q-row base

  const _Float16* qb = qd  + (size_t)bh * SEQ * HD;
  const _Float16* kb = kd  + (size_t)bh * SEQ * HD;
  const _Float16* vb = vTd + (size_t)bh * HD * SEQ;
  _Float16* pw = &plds[wave * 16 * LSTR];

  // Load Q fragments once, fold in softmax scale 1/sqrt(64) = 0.125 (exact in f16)
  f16x16 aq[2];
  {
    const size_t qrow = (size_t)(q0 + lmod) * HD;
#pragma unroll
    for (int dc = 0; dc < 2; ++dc) {
      const _Float16* p = qb + qrow + 32 * dc + 8 * lhalf;
      f16x16 a = cat8(*(const f16x8*)p, *(const f16x8*)(p + 16));
#pragma unroll
      for (int i = 0; i < 16; ++i) a[i] = a[i] * (_Float16)0.125f;
      aq[dc] = a;
    }
  }

  float mrow[8], lrow[8];
  f32x8 ctx[4];
#pragma unroll
  for (int r = 0; r < 8; ++r) { mrow[r] = -1e30f; lrow[r] = 0.0f; }
#pragma unroll
  for (int t = 0; t < 4; ++t) ctx[t] = f32x8{0,0,0,0,0,0,0,0};

  const int ntiles = (q0 + 16 + 31) >> 5;       // kv tiles needed for causal block
  for (int it = 0; it < ntiles; ++it) {
    const int kv0 = it * 32;

    // ---- scores: S = (Q*scale) . K^T  (2 n-tiles x 2 d-chunks = 4 WMMAs) ----
    f32x8 sc[2] = { f32x8{0,0,0,0,0,0,0,0}, f32x8{0,0,0,0,0,0,0,0} };
#pragma unroll
    for (int u = 0; u < 2; ++u) {
      const size_t krow = (size_t)(kv0 + 16 * u + lmod) * HD;
#pragma unroll
      for (int dc = 0; dc < 2; ++dc) {
        const _Float16* p = kb + krow + 32 * dc + 8 * lhalf;
        f16x16 bk = cat8(*(const f16x8*)p, *(const f16x8*)(p + 16));
        sc[u] = __builtin_amdgcn_wmma_f32_16x16x32_f16(
            false, aq[dc], false, bk, (short)0, sc[u], false, false);
      }
    }

    // ---- causal mask (boundary tiles only) ----
    if (kv0 + 31 > q0) {
#pragma unroll
      for (int u = 0; u < 2; ++u) {
        const int col = kv0 + 16 * u + lmod;
#pragma unroll
        for (int r = 0; r < 8; ++r) {
          const int row = q0 + 8 * lhalf + r;
          if (col > row) sc[u][r] = -1e30f;
        }
      }
    }

    // ---- online softmax: row reductions across the 16-lane half ----
    float rmax[8];
#pragma unroll
    for (int r = 0; r < 8; ++r) rmax[r] = fmaxf(sc[0][r], sc[1][r]);
#pragma unroll
    for (int off = 1; off < 16; off <<= 1)
#pragma unroll
      for (int r = 0; r < 8; ++r)
        rmax[r] = fmaxf(rmax[r], __shfl_xor(rmax[r], off, 32));

    float corr[8];
#pragma unroll
    for (int r = 0; r < 8; ++r) {
      const float mnew = fmaxf(mrow[r], rmax[r]);
      corr[r] = __expf(mrow[r] - mnew);
      mrow[r] = mnew;
    }
#pragma unroll
    for (int u = 0; u < 2; ++u)
#pragma unroll
      for (int r = 0; r < 8; ++r)
        sc[u][r] = __expf(sc[u][r] - mrow[r]);

    float rsum[8];
#pragma unroll
    for (int r = 0; r < 8; ++r) rsum[r] = sc[0][r] + sc[1][r];
#pragma unroll
    for (int off = 1; off < 16; off <<= 1)
#pragma unroll
      for (int r = 0; r < 8; ++r)
        rsum[r] += __shfl_xor(rsum[r], off, 32);
#pragma unroll
    for (int r = 0; r < 8; ++r) lrow[r] = lrow[r] * corr[r] + rsum[r];
#pragma unroll
    for (int t = 0; t < 4; ++t)
#pragma unroll
      for (int r = 0; r < 8; ++r) ctx[t][r] *= corr[r];

    // ---- P (C-layout) -> LDS -> A-layout fragment (cross-lane transpose) ----
#pragma unroll
    for (int u = 0; u < 2; ++u)
#pragma unroll
      for (int r = 0; r < 8; ++r)
        pw[(8 * lhalf + r) * LSTR + 16 * u + lmod] = (_Float16)sc[u][r];
    asm volatile("s_wait_dscnt 0x0" ::: "memory");   // wave-local LDS RAW ordering
    __builtin_amdgcn_wave_barrier();

    f16x16 pa;
    {
      const _Float16* p = pw + lmod * LSTR + 8 * lhalf;
      pa = cat8(*(const f16x8*)p, *(const f16x8*)(p + 16));
    }

    // ---- context += P . V   (4 n-tiles of d, K=32 kv) ----
#pragma unroll
    for (int t = 0; t < 4; ++t) {
      const _Float16* p = vb + (size_t)(16 * t + lmod) * SEQ + kv0 + 8 * lhalf;
      f16x16 bv = cat8(*(const f16x8*)p, *(const f16x8*)(p + 16));
      ctx[t] = __builtin_amdgcn_wmma_f32_16x16x32_f16(
          false, pa, false, bv, (short)0, ctx[t], false, false);
    }
  }

  // ---- normalize and store fp32 output [b][q][h*64+d] ----
#pragma unroll
  for (int t = 0; t < 4; ++t)
#pragma unroll
    for (int r = 0; r < 8; ++r) {
      const int row = q0 + 8 * lhalf + r;
      const float v = ctx[t][r] / lrow[r];
      out[((size_t)b * SEQ + row) * DIM + h * HD + 16 * t + lmod] = v;
    }
}

// ---------------------------------------------------------------------------
extern "C" void kernel_launch(void* const* d_in, const int* in_sizes, int n_in,
                              void* d_out, int out_size, void* d_ws, size_t ws_size,
                              hipStream_t stream) {
  const float* hidden = (const float*)d_in[0];
  const float* Wqkv   = (const float*)d_in[1];
  const float* bias   = (const float*)d_in[2];

  _Float16* q_ws  = (_Float16*)d_ws;                       // 8 MB
  _Float16* k_ws  = q_ws  + (size_t)NTOK * DIM;            // 8 MB
  _Float16* vT_ws = k_ws  + (size_t)NTOK * DIM;            // 8 MB  (24 MB total)

  dim3 g1(3 * DIM / 128, NTOK / 128);                      // (24, 32)
  qkv_rope_kernel<<<g1, 256, 0, stream>>>(hidden, Wqkv, bias, q_ws, k_ws, vT_ws);

  dim3 g2(4 * NH, SEQ / 64);                               // (64, 16)
  flash_attn_kernel<<<g2, 128, 0, stream>>>(q_ws, k_ws, vT_ws, (float*)d_out);
}
